// MixtureOfExperts_11123965297264
// MI455X (gfx1250) — matrix operand order repaired and checked by reading, not compiled, not run on hardware
//
#include <hip/hip_runtime.h>
#include <hip/hip_bf16.h>
#include <math.h>

typedef __bf16 bf16;
typedef __attribute__((ext_vector_type(16))) __bf16 v16bf;
typedef __attribute__((ext_vector_type(8)))  float  v8f;

#define E_ 8
#define D_ 512
#define H_ 1024
#define N_ 8192

__device__ __forceinline__ float gelu_exact(float v) {
  return 0.5f * v * (1.0f + erff(v * 0.70710678118654752f));
}

// -------------------------------------------------------------------------
// Router: one wave per token. logits -> softmax -> top2 -> renormalize.
// Writes probs (d_out tail), per-token combine weights, per-expert lists.
// -------------------------------------------------------------------------
__global__ __launch_bounds__(32) void moe_router(
    const float* __restrict__ x, const float* __restrict__ gw,
    const float* __restrict__ gb, float* __restrict__ probs_out,
    float* __restrict__ cw, int* __restrict__ counts, int* __restrict__ lists)
{
  const int n = blockIdx.x;
  const int lane = threadIdx.x;
  float acc[E_];
#pragma unroll
  for (int e = 0; e < E_; ++e) acc[e] = 0.f;

  const float* xr = x + (size_t)n * D_;
  for (int d = lane; d < D_; d += 32) {
    float xv = xr[d];
    const float4* g4 = (const float4*)(gw + (size_t)d * E_);
    float4 a = g4[0], b = g4[1];
    acc[0] += xv * a.x; acc[1] += xv * a.y; acc[2] += xv * a.z; acc[3] += xv * a.w;
    acc[4] += xv * b.x; acc[5] += xv * b.y; acc[6] += xv * b.z; acc[7] += xv * b.w;
  }
#pragma unroll
  for (int off = 16; off > 0; off >>= 1)
#pragma unroll
    for (int e = 0; e < E_; ++e) acc[e] += __shfl_xor(acc[e], off, 32);

#pragma unroll
  for (int e = 0; e < E_; ++e) acc[e] += gb[e];
  float mx = acc[0];
#pragma unroll
  for (int e = 1; e < E_; ++e) mx = fmaxf(mx, acc[e]);
  float p[E_]; float s = 0.f;
#pragma unroll
  for (int e = 0; e < E_; ++e) { p[e] = expf(acc[e] - mx); s += p[e]; }
  float inv = 1.f / s;
#pragma unroll
  for (int e = 0; e < E_; ++e) p[e] *= inv;

  int i0 = 0;
#pragma unroll
  for (int e = 1; e < E_; ++e) if (p[e] > p[i0]) i0 = e;
  int i1 = (i0 == 0) ? 1 : 0;
#pragma unroll
  for (int e = 0; e < E_; ++e) if (e != i0 && p[e] > p[i1]) i1 = e;
  float w0 = p[i0], w1 = p[i1];
  float iw = 1.f / (w0 + w1);

  if (lane == 0) {
    float* pr = probs_out + (size_t)n * E_;
#pragma unroll
    for (int e = 0; e < E_; ++e) pr[e] = p[e];
    cw[2 * n + 0] = w0 * iw;
    cw[2 * n + 1] = w1 * iw;
    int p0 = atomicAdd(&counts[i0], 1);
    lists[i0 * N_ + p0] = 2 * n + 0;           // slot id = token*2 + k
    int p1 = atomicAdd(&counts[i1], 1);
    lists[i1 * N_ + p1] = 2 * n + 1;
  }
}

// -------------------------------------------------------------------------
// One-time weight conversion fp32 -> bf16 hi/lo, stored in the exact WMMA
// B-operand per-lane layout (ISA 7.12.2): frag[e][ntile][kchunk][lane][16],
// element j of lane L corresponds to (K = (j<8 ? j : j+8) + 8*(L>>4), N = L&15).
// One wave per 32x16 fragment tile.
// -------------------------------------------------------------------------
__global__ __launch_bounds__(32) void prep_frags(
    const float* __restrict__ W, bf16* __restrict__ Fh, bf16* __restrict__ Fl,
    int Kd, int Nd)
{
  const int wid  = blockIdx.x;
  const int lane = threadIdx.x;
  const int kch = Kd >> 5, ntc = Nd >> 4;
  const int kc = wid % kch;
  const int nt = (wid / kch) % ntc;
  const int e  = wid / (kch * ntc);
  const float* Wp = W + (size_t)e * Kd * Nd;
  const int n = (nt << 4) + (lane & 15);
  const int half = lane >> 4;
  v16bf vh, vl;
#pragma unroll
  for (int j = 0; j < 16; ++j) {
    int k = (kc << 5) + (j < 8 ? j : j + 8) + (half << 3);
    float v = Wp[(size_t)k * Nd + n];
    bf16 hi = (bf16)v;
    vh[j] = hi;
    vl[j] = (bf16)(v - (float)hi);
  }
  size_t base = ((size_t)wid * 32 + lane) * 16;
  *(v16bf*)(Fh + base) = vh;
  *(v16bf*)(Fl + base) = vl;
}

// -------------------------------------------------------------------------
// Fused expert MLP: per block, 16 gathered tokens for one expert.
//   H1 = X @ W1 ; G = gelu(H1 + b1) ; Y = G @ W2 + b2
// bf16x3 split WMMA (hi*hi + hi*lo + lo*hi), fp32 accumulate.
// 8 waves: GEMM1 parallel over H chunks, G staged in LDS (A-layout),
// GEMM2 parallel over D tiles with persistent 16x512 fp32 accumulators.
// -------------------------------------------------------------------------
__global__ __launch_bounds__(256) void moe_expert_fused(
    const float* __restrict__ x,
    const float* __restrict__ b1, const float* __restrict__ b2,
    const bf16* __restrict__ w1fh, const bf16* __restrict__ w1fl,
    const bf16* __restrict__ w2fh, const bf16* __restrict__ w2fl,
    const int* __restrict__ counts, const int* __restrict__ lists,
    float* __restrict__ Y)
{
  __shared__ __align__(32) bf16 aXh[16][32][16];   // X tile, A-layout, hi
  __shared__ __align__(32) bf16 aXl[16][32][16];   // X tile, A-layout, lo
  __shared__ __align__(32) bf16 gH[8][32][16];     // G chunk, A-layout, hi
  __shared__ __align__(32) bf16 gL[8][32][16];     // G chunk, A-layout, lo
  __shared__ int sslot[16];

  const int e    = blockIdx.x >> 9;     // 512 token-tiles per expert
  const int tile = blockIdx.x & 511;
  const int m0   = tile << 4;
  const int cnt  = counts[e];
  if (m0 >= cnt) return;
  const int mcnt = min(16, cnt - m0);

  const int tid   = threadIdx.x;
  const int wv    = tid >> 5;
  const int lane  = tid & 31;
  const int lhalf = lane >> 4;
  const int lcol  = lane & 15;

  if (tid < 16)
    sslot[tid] = (tid < mcnt) ? lists[e * N_ + m0 + tid] : 0;
  __syncthreads();

  // Gather X rows -> bf16 hi/lo into LDS in WMMA A-layout:
  // element j of lane L (m = L&15) holds K = (j<8 ? j : j+8) + 8*(L>>4).
#pragma unroll
  for (int it = 0; it < 32; ++it) {
    int idx = it * 256 + tid;             // [16 rows][512 cols]
    int m = idx >> 9;
    int k = idx & 511;
    float v = 0.f;
    if (m < mcnt) v = x[(size_t)(sslot[m] >> 1) * D_ + k];
    bf16 hi = (bf16)v;
    bf16 lo = (bf16)(v - (float)hi);
    int c = k >> 5, r = k & 31;
    int half = (r >> 3) & 1;
    int j = (r & 7) + ((r >> 4) << 3);
    aXh[c][m + (half << 4)][j] = hi;
    aXl[c][m + (half << 4)][j] = lo;
  }
  __syncthreads();

  v8f yacc[4];
#pragma unroll
  for (int u = 0; u < 4; ++u)
#pragma unroll
    for (int v = 0; v < 8; ++v) yacc[u][v] = 0.f;

  for (int hb = 0; hb < 4; ++hb) {        // H processed in blocks of 256
    // ---- GEMM1: wave wv computes H1 for h in [hb*256 + wv*32, +32) ----
#pragma unroll
    for (int t = 0; t < 2; ++t) {
      v8f hacc;
#pragma unroll
      for (int v = 0; v < 8; ++v) hacc[v] = 0.f;
      int ntile1 = (hb << 4) + (wv << 1) + t;           // 0..63
      const bf16* bph = w1fh + (((size_t)e * 64 + ntile1) * 16) * 512 + (size_t)lane * 16;
      const bf16* bpl = w1fl + (((size_t)e * 64 + ntile1) * 16) * 512 + (size_t)lane * 16;
#pragma unroll
      for (int kc = 0; kc < 16; ++kc) {
        v16bf ah = *(const v16bf*)&aXh[kc][lane][0];
        v16bf al = *(const v16bf*)&aXl[kc][lane][0];
        v16bf bh = *(const v16bf*)(bph + (size_t)kc * 512);
        v16bf bl = *(const v16bf*)(bpl + (size_t)kc * 512);
        hacc = __builtin_amdgcn_wmma_f32_16x16x32_bf16(false, ah, false, bh, (short)0, hacc, false, false);
        hacc = __builtin_amdgcn_wmma_f32_16x16x32_bf16(false, ah, false, bl, (short)0, hacc, false, false);
        hacc = __builtin_amdgcn_wmma_f32_16x16x32_bf16(false, al, false, bh, (short)0, hacc, false, false);
      }
      // bias + exact GELU, re-split to bf16 hi/lo, store as GEMM2 A-frag
      int hcol = (hb << 8) + (wv << 5) + (t << 4) + lcol;
      float b1v = b1[(size_t)e * H_ + hcol];
      int r = (t << 4) + lcol;
      int half = (r >> 3) & 1;
      int j = (r & 7) + ((r >> 4) << 3);
#pragma unroll
      for (int v = 0; v < 8; ++v) {
        int m = v + (lhalf << 3);
        float g = gelu_exact(hacc[v] + b1v);
        bf16 gh = (bf16)g;
        gH[wv][m + (half << 4)][j] = gh;
        gL[wv][m + (half << 4)][j] = (bf16)(g - (float)gh);
      }
    }
    __syncthreads();

    // ---- GEMM2: Y[16,512] += G[16,256-block] @ W2[block,512] ----
#pragma unroll
    for (int kc = 0; kc < 8; ++kc) {
      v16bf ah = *(const v16bf*)&gH[kc][lane][0];
      v16bf al = *(const v16bf*)&gL[kc][lane][0];
      int kcg = (hb << 3) + kc;                         // 0..31
#pragma unroll
      for (int u = 0; u < 4; ++u) {
        int ntile2 = (wv << 2) + u;                     // 0..31
        size_t fidx = ((((size_t)e * 32 + ntile2) * 32 + kcg) * 32 + lane) * 16;
        v16bf bh = *(const v16bf*)(w2fh + fidx);
        v16bf bl = *(const v16bf*)(w2fl + fidx);
        yacc[u] = __builtin_amdgcn_wmma_f32_16x16x32_bf16(false, ah, false, bh, (short)0, yacc[u], false, false);
        yacc[u] = __builtin_amdgcn_wmma_f32_16x16x32_bf16(false, ah, false, bl, (short)0, yacc[u], false, false);
        yacc[u] = __builtin_amdgcn_wmma_f32_16x16x32_bf16(false, al, false, bh, (short)0, yacc[u], false, false);
      }
    }
    __syncthreads();
  }

  // ---- add b2, scatter Y per assignment slot ----
#pragma unroll
  for (int u = 0; u < 4; ++u) {
    int col = (wv << 6) + (u << 4) + lcol;
    float b2v = b2[(size_t)e * D_ + col];
#pragma unroll
    for (int v = 0; v < 8; ++v) {
      int m = v + (lhalf << 3);
      if (m < mcnt)
        Y[(size_t)sslot[m] * D_ + col] = yacc[u][v] + b2v;
    }
  }
}

// -------------------------------------------------------------------------
// Final combine: out[n] = w0 * Y[2n] + w1 * Y[2n+1]
// -------------------------------------------------------------------------
__global__ __launch_bounds__(256) void moe_combine(
    const float* __restrict__ Y, const float* __restrict__ cw,
    float* __restrict__ out)
{
  size_t i = (size_t)blockIdx.x * 256 + threadIdx.x;
  if (i >= (size_t)N_ * D_) return;
  size_t n = i / D_;
  size_t d = i - n * D_;
  out[i] = cw[2 * n]     * Y[(2 * n)     * D_ + d]
         + cw[2 * n + 1] * Y[(2 * n + 1) * D_ + d];
}

extern "C" void kernel_launch(void* const* d_in, const int* in_sizes, int n_in,
                              void* d_out, int out_size, void* d_ws, size_t ws_size,
                              hipStream_t stream) {
  const float* x  = (const float*)d_in[0];
  const float* gw = (const float*)d_in[1];
  const float* gb = (const float*)d_in[2];
  const float* w1 = (const float*)d_in[3];
  const float* b1 = (const float*)d_in[4];
  const float* w2 = (const float*)d_in[5];
  const float* b2 = (const float*)d_in[6];

  float* out_y  = (float*)d_out;                     // [N, D]
  float* out_pr = (float*)d_out + (size_t)N_ * D_;   // [N, E]

  // Workspace carve-up (~67.5 MB), all sub-buffers 32B-aligned.
  char* p = (char*)d_ws;
  int*   counts = (int*)p;   p += 256;
  int*   lists  = (int*)p;   p += (size_t)E_ * N_ * sizeof(int);
  float* cw     = (float*)p; p += (size_t)N_ * 2 * sizeof(float);
  bf16*  w1fh   = (bf16*)p;  p += (size_t)E_ * D_ * H_ * sizeof(bf16);
  bf16*  w1fl   = (bf16*)p;  p += (size_t)E_ * D_ * H_ * sizeof(bf16);
  bf16*  w2fh   = (bf16*)p;  p += (size_t)E_ * H_ * D_ * sizeof(bf16);
  bf16*  w2fl   = (bf16*)p;  p += (size_t)E_ * H_ * D_ * sizeof(bf16);
  float* Y      = (float*)p; p += (size_t)N_ * 2 * D_ * sizeof(float);

  hipMemsetAsync(counts, 0, E_ * sizeof(int), stream);
  moe_router<<<N_, 32, 0, stream>>>(x, gw, gb, out_pr, cw, counts, lists);
  prep_frags<<<E_ * 64 * 16, 32, 0, stream>>>(w1, w1fh, w1fl, D_, H_);
  prep_frags<<<E_ * 32 * 32, 32, 0, stream>>>(w2, w2fh, w2fl, H_, D_);
  moe_expert_fused<<<E_ * 512, 256, 0, stream>>>(x, b1, b2, w1fh, w1fl,
                                                 w2fh, w2fl, counts, lists, Y);
  moe_combine<<<(int)(((size_t)N_ * D_ + 255) / 256), 256, 0, stream>>>(Y, cw, out_y);
}
